// Metric_Loss_2654289789012
// MI455X (gfx1250) — matrix operand order repaired and checked by reading, not compiled, not run on hardware
//
#include <hip/hip_runtime.h>
#include <hip/hip_bf16.h>

typedef __attribute__((ext_vector_type(16))) _Float16 v16h;
typedef __attribute__((ext_vector_type(8)))  float    v8f;

#define NROWS 8192
#define NFEAT 128
#define NPAIR 4096
#define NTILE (NROWS / 16)   // 512 column tiles of 16

// ---------------------------------------------------------------------------
// Kernel 1: build f16 copies of X (text) and Y (interleaved text/shape),
// zero the row-sum scratch and the output scalar.
// ---------------------------------------------------------------------------
__global__ __launch_bounds__(256) void ml_prep(const float* __restrict__ X,
                                               const float* __restrict__ Sh,
                                               _Float16* __restrict__ Xh,
                                               _Float16* __restrict__ Yh,
                                               float* __restrict__ S,
                                               float* __restrict__ out) {
  int idx = blockIdx.x * blockDim.x + threadIdx.x;      // 0 .. 8192*128-1
  int row = idx >> 7;
  int col = idx & (NFEAT - 1);
  float xv = X[idx];
  Xh[idx] = (_Float16)xv;
  // Y: even rows = text rows, odd rows 2p+1 = shape[p]
  float yv = (row & 1) ? Sh[(row >> 1) * NFEAT + col] : xv;
  Yh[idx] = (_Float16)yv;
  if (idx < 2 * NROWS) S[idx] = 0.0f;
  if (idx == 0) out[0] = 0.0f;
}

// ---------------------------------------------------------------------------
// B-fragment loader: B[k][n] = M[col_n][k], K contiguous in memory.
// lanes 0-15 : col = l16, K = k0+{0..7,16..23}; lanes 16-31 : K = k0+{8..15,24..31}
// ---------------------------------------------------------------------------
__device__ __forceinline__ void ml_load_btile(const _Float16* __restrict__ M,
                                              int ct, int l16, int sel,
                                              v16h b[4]) {
  const _Float16* cp = M + (ct * 16 + l16) * NFEAT;
#pragma unroll
  for (int q = 0; q < 4; ++q) {
    int k0 = q * 32 + sel * 8;
    union { v16h h; uint4 u[2]; } t;
    t.u[0] = *(const uint4*)(cp + k0);
    t.u[1] = *(const uint4*)(cp + k0 + 16);
    b[q] = t.h;
  }
}

// ---------------------------------------------------------------------------
// Kernel 2: row sums of exp(1 + (M M^T)/128) using v_wmma_f32_16x16x32_f16.
// grid = (512 row tiles, 2 matrices), block = 256 threads = 8 waves.
// Wave w handles column tiles w, w+8, ... with double-buffered B fragments
// so the next tile's 8 global_load_b128 overlap the current tile's WMMAs+exp.
// ---------------------------------------------------------------------------
__global__ __launch_bounds__(256, 1) void ml_rowsum(const _Float16* __restrict__ Xh,
                                                    const _Float16* __restrict__ Yh,
                                                    float* __restrict__ S) {
  const _Float16* __restrict__ M = (blockIdx.y == 0) ? Xh : Yh;
  float* __restrict__ Srow = S + blockIdx.y * NROWS;

  const int wave = threadIdx.x >> 5;
  const int lane = threadIdx.x & 31;
  const int sel  = lane >> 4;        // which 16-lane half
  const int l16  = lane & 15;
  const int RA   = blockIdx.x * 16;  // row-tile base

  // ---- Load A fragments once: 16 rows x 128 K, ISA 16-bit A layout ----
  // lanes 0-15  : M = lane,    K = k0+{0..7}  (v0-3), k0+{16..23} (v4-7)
  // lanes 16-31 : M = lane-16, K = k0+{8..15} (v0-3), k0+{24..31} (v4-7)
  v16h a[4];
  {
    const _Float16* rp = M + (RA + l16) * NFEAT;
#pragma unroll
    for (int q = 0; q < 4; ++q) {
      int k0 = q * 32 + sel * 8;
      union { v16h h; uint4 u[2]; } t;
      t.u[0] = *(const uint4*)(rp + k0);
      t.u[1] = *(const uint4*)(rp + k0 + 16);
      a[q] = t.h;
    }
  }

  v8f acc = {0.f, 0.f, 0.f, 0.f, 0.f, 0.f, 0.f, 0.f};

  // pipeline prologue: B fragments for this wave's first tile
  v16h bc[4];
  ml_load_btile(M, wave, l16, sel, bc);

  for (int ct = wave; ct < NTILE; ct += 8) {
    // issue next tile's loads (branch-free: clamp to a valid tile on the
    // final iteration; the loaded data is simply discarded)
    int nt = (ct + 8 < NTILE) ? (ct + 8) : wave;
    v16h bn[4];
    ml_load_btile(M, nt, l16, sel, bn);

    // hint two tiles ahead into cache (global_prefetch_b8), branch-free
    int pt = (ct + 16 < NTILE) ? (ct + 16) : wave;
    __builtin_prefetch(M + (pt * 16 + l16) * NFEAT, 0, 3);

    // current tile: 4 back-to-back WMMAs over registers already resident
    v8f c = {0.f, 0.f, 0.f, 0.f, 0.f, 0.f, 0.f, 0.f};
#pragma unroll
    for (int q = 0; q < 4; ++q) {
      c = __builtin_amdgcn_wmma_f32_16x16x32_f16(false, a[q], false, bc[q],
                                                 (short)0, c, false, false);
    }
    // c = 16x16 Gram tile; E = exp(1 + c/128); accumulate per-lane partials
#pragma unroll
    for (int v = 0; v < 8; ++v) {
      acc[v] += __expf(1.0f + c[v] * 0.0078125f);
    }
#pragma unroll
    for (int q = 0; q < 4; ++q) bc[q] = bn[q];
  }

  // C layout: VGPR v, lanes 0-15 -> row RA+v; lanes 16-31 -> row RA+8+v.
  // Butterfly-reduce within each 16-lane half, then one atomic per row.
#pragma unroll
  for (int v = 0; v < 8; ++v) {
    float x = acc[v];
    x += __shfl_xor(x, 1);
    x += __shfl_xor(x, 2);
    x += __shfl_xor(x, 4);
    x += __shfl_xor(x, 8);
    if (l16 == 0) atomicAdd(&Srow[RA + sel * 8 + v], x);
  }
}

// ---------------------------------------------------------------------------
// Kernel 3: per-pair terms in fp32 + final scalar reduction.
// t in [0, 8192): matrix m = t>>12, pair p = t&4095.
// ---------------------------------------------------------------------------
__global__ __launch_bounds__(256) void ml_pairloss(const float* __restrict__ X,
                                                   const float* __restrict__ Sh,
                                                   const float* __restrict__ S,
                                                   float* __restrict__ out) {
  int t = blockIdx.x * blockDim.x + threadIdx.x;
  int m = t >> 12;
  int p = t & (NPAIR - 1);
  int i = 2 * p, j = 2 * p + 1;
  const float* ri = X + i * NFEAT;                       // both matrices: even row = text
  const float* rj = (m == 0) ? (X + j * NFEAT) : (Sh + p * NFEAT);

  float dii = 0.f, dij = 0.f, djj = 0.f;
#pragma unroll 8
  for (int k = 0; k < NFEAT; k += 4) {
    float4 va = *(const float4*)(ri + k);
    float4 vb = *(const float4*)(rj + k);
    dii += va.x * va.x + va.y * va.y + va.z * va.z + va.w * va.w;
    dij += va.x * vb.x + va.y * vb.y + va.z * vb.z + va.w * vb.w;
    djj += vb.x * vb.x + vb.y * vb.y + vb.z * vb.z + vb.w * vb.w;
  }
  const float inv = 0.0078125f;                          // 1/128
  float Dii = dii * inv, Dij = dij * inv, Djj = djj * inv;
  float Eii = __expf(1.f + Dii);
  float Eij = __expf(1.f + Dij);
  float Ejj = __expf(1.f + Djj);
  float Si = S[m * NROWS + i], Sj = S[m * NROWS + j];
  float neg = (Si + Sj) - (Eii + 2.f * Eij + Ejj);
  float J = __logf(neg) - Dij;
  float r = fmaxf(J, 0.f);
  // mean over P of relu^2/(2P)  ==  sum relu^2 / (2 P^2)
  float contrib = r * r * (1.0f / (2.0f * (float)NPAIR * (float)NPAIR));

  // wave reduction, one atomic per wave
#pragma unroll
  for (int msk = 1; msk < 32; msk <<= 1) contrib += __shfl_xor(contrib, msk);
  if ((threadIdx.x & 31) == 0) atomicAdd(out, contrib);
}

// ---------------------------------------------------------------------------
extern "C" void kernel_launch(void* const* d_in, const int* in_sizes, int n_in,
                              void* d_out, int out_size, void* d_ws, size_t ws_size,
                              hipStream_t stream) {
  const float* X  = (const float*)d_in[0];   // [8192,128] f32
  const float* Sh = (const float*)d_in[1];   // [4096,128] f32
  float* out = (float*)d_out;

  // workspace layout: Xh (2MB f16) | Yh (2MB f16) | S (2*8192 f32)
  _Float16* Xh = (_Float16*)d_ws;
  _Float16* Yh = Xh + (size_t)NROWS * NFEAT;
  float*    S  = (float*)(Yh + (size_t)NROWS * NFEAT);

  ml_prep<<<(NROWS * NFEAT) / 256, 256, 0, stream>>>(X, Sh, Xh, Yh, S, out);

  dim3 grid(NTILE, 2);
  ml_rowsum<<<grid, 256, 0, stream>>>(Xh, Yh, S);

  ml_pairloss<<<(2 * NPAIR) / 256, 256, 0, stream>>>(X, Sh, S, out);
}
